// SelfCorrelationPercPooling_74534862454957
// MI455X (gfx1250) — matrix operand order repaired and checked by reading, not compiled, not run on hardware
//
#include <hip/hip_runtime.h>

typedef __attribute__((ext_vector_type(16))) __bf16 v16bf;
typedef __attribute__((ext_vector_type(8)))  __bf16 v8bf;
typedef __attribute__((ext_vector_type(8)))  float  v8f;

#define BATCH    32
#define NMAPS    1024
#define FDIM     512
#define NB_POOLS 256

// ---------------------------------------------------------------------------
// Kernel 1: split fp32 x into bf16 hi + bf16 lo residual (for 3-pass WMMA).
// ---------------------------------------------------------------------------
__global__ __launch_bounds__(256) void scpp_split_bf16(
    const float* __restrict__ x, __bf16* __restrict__ xhi,
    __bf16* __restrict__ xlo, int n) {
  int i = blockIdx.x * blockDim.x + threadIdx.x;
  if (i < n) {
    float v = x[i];
    __bf16 h = (__bf16)v;          // RTNE f32->bf16
    xhi[i] = h;
    xlo[i] = (__bf16)(v - (float)h);
  }
}

// ---------------------------------------------------------------------------
// Kernel 2: fused Gram-matrix (WMMA bf16 hi/lo split) + bitonic sort + gather.
// One block = one (batch b, 16-row m-stripe). 8 waves x 8 n-tiles = full
// 16x1024 corr stripe in LDS (64 KB), never touches global memory.
// ---------------------------------------------------------------------------
__global__ __launch_bounds__(256) void scpp_corr_sort(
    const __bf16* __restrict__ xhi, const __bf16* __restrict__ xlo,
    float* __restrict__ out) {
  __shared__ float stripe[16][NMAPS];   // 64 KB

  const int tid  = threadIdx.x;
  const int lane = tid & 31;
  const int wave = tid >> 5;            // 0..7
  const int m0   = blockIdx.x * 16;     // m-stripe base
  const int b    = blockIdx.y;          // batch

  const __bf16* Xh = xhi + (size_t)b * NMAPS * FDIM;
  const __bf16* Xl = xlo + (size_t)b * NMAPS * FDIM;

  const int l16 = lane & 15;
  const int lhi = lane >> 4;            // 0 or 1 (half-wave)

  // A fragment (16x32 bf16): lanes 0-15 hold row M=lane, contiguous K
  // [k0..k0+15]; lanes 16-31 shift K base by +16.
  const __bf16* Ah_base = Xh + (size_t)(m0 + l16) * FDIM + lhi * 16;
  const __bf16* Al_base = Xl + (size_t)(m0 + l16) * FDIM + lhi * 16;

  v8f acc[8] = {};   // 8 n-tiles of 16x16 f32 accumulators

  for (int k0 = 0; k0 < FDIM; k0 += 32) {
    v8bf ah0 = *(const v8bf*)(Ah_base + k0);
    v8bf ah1 = *(const v8bf*)(Ah_base + k0 + 8);
    v8bf al0 = *(const v8bf*)(Al_base + k0);
    v8bf al1 = *(const v8bf*)(Al_base + k0 + 8);
    v16bf Ahi = __builtin_shufflevector(ah0, ah1,
        0,1,2,3,4,5,6,7,8,9,10,11,12,13,14,15);
    v16bf Alo = __builtin_shufflevector(al0, al1,
        0,1,2,3,4,5,6,7,8,9,10,11,12,13,14,15);

    // B fragment (32x16 bf16): lane holds column N = n0+l16, contiguous
    // K range [kb, kb+16) (32 bytes), upper half-wave takes K+16.
    const int kb = k0 + lhi * 16;

#pragma unroll
    for (int t = 0; t < 8; ++t) {
      const int n0 = (wave * 8 + t) * 16;
      const size_t boff = (size_t)(n0 + l16) * FDIM + kb;
      v16bf Bhi = *(const v16bf*)(Xh + boff);
      v16bf Blo = *(const v16bf*)(Xl + boff);
      // corr ~= hi*hi^T + hi*lo^T + lo*hi^T  (drop lo*lo, ~2^-16 rel)
      acc[t] = __builtin_amdgcn_wmma_f32_16x16x32_bf16(
          false, Ahi, false, Bhi, (short)0, acc[t], false, false);
      acc[t] = __builtin_amdgcn_wmma_f32_16x16x32_bf16(
          false, Ahi, false, Blo, (short)0, acc[t], false, false);
      acc[t] = __builtin_amdgcn_wmma_f32_16x16x32_bf16(
          false, Alo, false, Bhi, (short)0, acc[t], false, false);
    }
  }

  // Spill accumulators into the LDS stripe using the documented C/D layout:
  // element v of acc -> M = v + 8*lhi, N = n0 + l16.
  const float scale = 1.0f / (float)FDIM;
  const int mrow = lhi * 8;
#pragma unroll
  for (int t = 0; t < 8; ++t) {
    const int n = (wave * 8 + t) * 16 + l16;
#pragma unroll
    for (int v = 0; v < 8; ++v) {
      stripe[mrow + v][n] = acc[t][v] * scale;
    }
  }
  __syncthreads();

  // Bitonic ascending sort of each of the 16 rows (1024 elements each),
  // all rows swept in parallel per stage.
  for (int k = 2; k <= NMAPS; k <<= 1) {
    for (int j = k >> 1; j > 0; j >>= 1) {
      for (int p = tid; p < 16 * NMAPS; p += 256) {
        const int row = p >> 10;
        const int i   = p & (NMAPS - 1);
        const int ixj = i ^ j;
        if (ixj > i) {
          float a = stripe[row][i];
          float c = stripe[row][ixj];
          const bool up = ((i & k) == 0);
          if ((a > c) == up) { stripe[row][i] = c; stripe[row][ixj] = a; }
        }
      }
      __syncthreads();
    }
  }

  // Gather percentile ranks: ranks = round(linspace(1, 1023, 256));
  // descending-sorted[r] == ascending-sorted[1023 - r].
  float* outb = out + ((size_t)b * NMAPS + m0) * NB_POOLS;
  for (int p = tid; p < 16 * NB_POOLS; p += 256) {
    const int row = p >> 8;
    const int i   = p & (NB_POOLS - 1);
    const float rf = 1.0f + (float)i * (1022.0f / 255.0f);
    const int rank = (int)rintf(rf);         // RTNE matches jnp.round
    outb[row * NB_POOLS + i] = stripe[row][(NMAPS - 1) - rank];
  }
}

// ---------------------------------------------------------------------------
extern "C" void kernel_launch(void* const* d_in, const int* in_sizes, int n_in,
                              void* d_out, int out_size, void* d_ws, size_t ws_size,
                              hipStream_t stream) {
  const float* x = (const float*)d_in[0];
  float* out = (float*)d_out;

  const int n = BATCH * NMAPS * FDIM;           // 16,777,216 elements
  __bf16* xhi = (__bf16*)d_ws;                  // 33.5 MB
  __bf16* xlo = xhi + (size_t)n;                // 33.5 MB (ws total ~67 MB)

  scpp_split_bf16<<<(n + 255) / 256, 256, 0, stream>>>(x, xhi, xlo, n);

  dim3 grid(NMAPS / 16, BATCH);                 // 64 m-stripes x 32 batches
  scpp_corr_sort<<<grid, 256, 0, stream>>>(xhi, xlo, out);
}
